// BiLSTM_CRF_50173807952498
// MI455X (gfx1250) — matrix operand order repaired and checked by reading, not compiled, not run on hardware
//
#include <hip/hip_runtime.h>

// ---------------------------------------------------------------------------
// BiLSTM-CRF for MI455X (gfx1250, wave32).
//   1) prep:   gather X = emb[sen] -> bf16, convert Wih -> bf16, bias sums
//   2) gemm:   P[dir][t][g] = X @ Wih^T + (bih+bhh)          (WMMA bf16)
//   3) lstm:   recurrence with Whh held persistently in VGPRs as WMMA A-tiles
//   4) feats:  feats[t][k] = [hf|hb][t] . Wout[k] + bout[k]
//   5) viterbi: single-wave scan + backtrace
// ---------------------------------------------------------------------------

#define T_LEN 4096
#define EMB   256
#define H2    256
#define G4    1024     // 4*H2 gate rows
#define TAGS  5
#define START_TAG 3
#define STOP_TAG  4
#define NEGV  (-10000.0f)

typedef __bf16 bf16_t;
typedef __attribute__((ext_vector_type(16))) __bf16 v16bf;
typedef __attribute__((ext_vector_type(8)))  __bf16 v8bf;
typedef __attribute__((ext_vector_type(8)))  float  v8f;

// ------------------------------- phase 1 -----------------------------------
__global__ void __launch_bounds__(256)
prep_kernel(const int* __restrict__ sen, const float* __restrict__ emb,
            const float* __restrict__ Wih_f, const float* __restrict__ Wih_b,
            const float* __restrict__ bih_f, const float* __restrict__ bhh_f,
            const float* __restrict__ bih_b, const float* __restrict__ bhh_b,
            bf16_t* __restrict__ Xbf, bf16_t* __restrict__ Wbf,
            float* __restrict__ bsum) {
  int b = blockIdx.x, e = threadIdx.x;
  if (b < T_LEN) {
    int tok = sen[b];
    Xbf[(size_t)b * EMB + e] = (bf16_t)emb[(size_t)tok * EMB + e];
  } else if (b < T_LEN + 2 * G4) {
    int q = b - T_LEN;
    int dir = q >> 10;
    int r   = q & (G4 - 1);
    const float* W = dir ? Wih_b : Wih_f;
    Wbf[((size_t)dir * G4 + r) * EMB + e] = (bf16_t)W[(size_t)r * EMB + e];
  } else {
    int idx = (b - (T_LEN + 2 * G4)) * 256 + e;   // 0 .. 2047
    if (idx < 2 * G4) {
      int dir = idx >> 10;
      int g   = idx & (G4 - 1);
      bsum[idx] = dir ? (bih_b[g] + bhh_b[g]) : (bih_f[g] + bhh_f[g]);
    }
  }
}

// ------------------------------- phase 2 (WMMA GEMM) -----------------------
// A (16x32 bf16): lane l<16 row M=l holds K 0..15; lanes 16..31 hold
//                 K 16..31  -> two aligned 16-byte loads per lane.
// B (32x16 bf16): lane l col N=l&15; lanes 0-15 K=0..15, lanes 16-31 K=16..31
//                 -> one aligned 32-byte load per lane (Wih row-major over K).
// C/D (16x16 f32): lane l col=l&15, VGPR r row = r + (l>=16 ? 8 : 0).
__global__ void __launch_bounds__(128)
input_gemm_kernel(const bf16_t* __restrict__ Xbf, const bf16_t* __restrict__ Wbf,
                  const float* __restrict__ bsum, float* __restrict__ P) {
  const int lane = threadIdx.x & 31;
  const int wave = threadIdx.x >> 5;
  const int mt   = blockIdx.x;                // 0..255  (T/16)
  const int nt   = blockIdx.y * 4 + wave;     // 0..63   (G4/16)
  const int dir  = blockIdx.z;                // 0..1

  const int arow = mt * 16 + (lane & 15);
  const int hi   = (lane & 16) ? 8 : 0;       // D row offset
  const int gcol = nt * 16 + (lane & 15);
  const int boff = (lane & 16) ? 16 : 0;

  const bf16_t* Abase = Xbf + (size_t)arow * EMB;
  const bf16_t* Bbase = Wbf + ((size_t)dir * G4 + gcol) * EMB;

  v8f acc = {};
#pragma unroll
  for (int kt = 0; kt < EMB / 32; ++kt) {
    const int k0 = kt * 32;
    v8bf a_lo = *(const v8bf*)(Abase + k0 + boff);
    v8bf a_hi = *(const v8bf*)(Abase + k0 + boff + 8);
    v16bf a = __builtin_shufflevector(a_lo, a_hi,
                                      0, 1, 2, 3, 4, 5, 6, 7,
                                      8, 9, 10, 11, 12, 13, 14, 15);
    v16bf bmat = *(const v16bf*)(Bbase + k0 + boff);
    acc = __builtin_amdgcn_wmma_f32_16x16x32_bf16(
        false, a, false, bmat, (short)0, acc, false, false);
  }

  const float bias  = bsum[dir * G4 + gcol];
  const int   trow0 = mt * 16 + hi;
  const size_t base = (size_t)dir * T_LEN * G4;
#pragma unroll
  for (int r = 0; r < 8; ++r)
    P[base + (size_t)(trow0 + r) * G4 + gcol] = acc[r] + bias;
}

// ------------------------------- phase 3 (WMMA recurrence) -----------------
__device__ __forceinline__ float sigm(float x) {
  return 1.0f / (1.0f + __expf(-x));
}

// One block per direction, 32 waves. Wave w holds gate rows [32w, 32w+32) of
// Whh as 2 M-tiles x 8 K-tiles of bf16 in VGPRs (128 VGPRs/lane), persistent
// across all T steps: zero steady-state weight traffic.
// Per step: B = h broadcast into all 16 columns (LDS, 32B/lane), 16 chained
// WMMAs (2 independent accumulator chains), lanes 0/16 extract column 0 of D
// into the gate-preact LDS buffer, threads 0..255 do the pointwise update.
__global__ void __launch_bounds__(1024)
lstm_wmma_kernel(const float* __restrict__ P, const float* __restrict__ Whh_f,
                 const float* __restrict__ Whh_b, const float* __restrict__ h0,
                 const float* __restrict__ c0, float* __restrict__ hcat) {
  __shared__ __align__(32) bf16_t hbf[H2];
  __shared__ __align__(16) float  gpre[G4];

  const int dir  = blockIdx.x;
  const int tid  = threadIdx.x;
  const int lane = tid & 31;
  const int wv   = tid >> 5;                  // 0..31
  const float* Whh = dir ? Whh_b : Whh_f;

  // ---- load persistent A tiles (convert f32 -> bf16 once) -------------
  // Explicit float4 quads -> guaranteed global_load_b128, then pack.
  const int arow  = wv * 32 + (lane & 15);    // tile 0 row for this lane
  const int kb_lo = (lane & 16) ? 16 : 0;     // A K sub-offset per layout
  v16bf A[2][8];
#pragma unroll
  for (int m = 0; m < 2; ++m) {
    const float* rp = Whh + (size_t)(arow + m * 16) * H2;
#pragma unroll
    for (int kt = 0; kt < 8; ++kt) {
      const float4* q0 = (const float4*)(rp + kt * 32 + kb_lo);
      const float4* q1 = (const float4*)(rp + kt * 32 + 8 + kb_lo);
      float4 w0 = q0[0], w1 = q0[1];          // K: base+0..7
      float4 w2 = q1[0], w3 = q1[1];          // K: base+8..15
      v16bf a;
      a[0]  = (bf16_t)w0.x; a[1]  = (bf16_t)w0.y; a[2]  = (bf16_t)w0.z; a[3]  = (bf16_t)w0.w;
      a[4]  = (bf16_t)w1.x; a[5]  = (bf16_t)w1.y; a[6]  = (bf16_t)w1.z; a[7]  = (bf16_t)w1.w;
      a[8]  = (bf16_t)w2.x; a[9]  = (bf16_t)w2.y; a[10] = (bf16_t)w2.z; a[11] = (bf16_t)w2.w;
      a[12] = (bf16_t)w3.x; a[13] = (bf16_t)w3.y; a[14] = (bf16_t)w3.z; a[15] = (bf16_t)w3.w;
      A[m][kt] = a;
    }
  }

  float c = 0.0f;
  if (tid < H2) {
    hbf[tid] = (bf16_t)h0[dir * H2 + tid];
    c        = c0[dir * H2 + tid];
  }
  __syncthreads();

  const float* Pd    = P + (size_t)dir * T_LEN * G4;
  const int    koff  = (lane & 16) ? 16 : 0;              // B K sub-offset
  const int    gbase = wv * 32 + ((lane & 16) ? 8 : 0);   // D row base (col 0)
  float* hout = hcat + dir * H2 + tid;                    // valid for tid<H2

  for (int s = 0; s < T_LEN; ++s) {
    const int t = dir ? (T_LEN - 1 - s) : s;
    const float* Pt = Pd + (size_t)t * G4;

    // P prefetch for the pointwise stage: issued before (hidden under) WMMAs
    float p_i = 0.f, p_f = 0.f, p_g = 0.f, p_o = 0.f;
    if (tid < H2) {
      p_i = Pt[tid];
      p_f = Pt[H2 + tid];
      p_g = Pt[2 * H2 + tid];
      p_o = Pt[3 * H2 + tid];
    }

    v8f acc0 = {}, acc1 = {};
#pragma unroll
    for (int kt = 0; kt < 8; ++kt) {
      // h replicated into all 16 B columns: element j = h[kt*32 + koff + j]
      v16bf bmat = *(const v16bf*)(hbf + kt * 32 + koff);
      acc0 = __builtin_amdgcn_wmma_f32_16x16x32_bf16(
          false, A[0][kt], false, bmat, (short)0, acc0, false, false);
      acc1 = __builtin_amdgcn_wmma_f32_16x16x32_bf16(
          false, A[1][kt], false, bmat, (short)0, acc1, false, false);
    }

    // Column 0 of D lives in lanes 0 (rows r) and 16 (rows 8+r).
    if ((lane & 15) == 0) {
      *(float4*)(gpre + gbase)      = make_float4(acc0[0], acc0[1], acc0[2], acc0[3]);
      *(float4*)(gpre + gbase + 4)  = make_float4(acc0[4], acc0[5], acc0[6], acc0[7]);
      *(float4*)(gpre + gbase + 16) = make_float4(acc1[0], acc1[1], acc1[2], acc1[3]);
      *(float4*)(gpre + gbase + 20) = make_float4(acc1[4], acc1[5], acc1[6], acc1[7]);
    }
    __syncthreads();

    if (tid < H2) {
      const float ig = sigm(gpre[tid] + p_i);
      const float fg = sigm(gpre[H2 + tid] + p_f);
      const float gg = tanhf(gpre[2 * H2 + tid] + p_g);
      const float og = sigm(gpre[3 * H2 + tid] + p_o);
      c = fg * c + ig * gg;
      const float h = og * tanhf(c);
      hout[(size_t)t * (2 * H2)] = h;
      hbf[tid] = (bf16_t)h;
    }
    __syncthreads();
  }
}

// ------------------------------- phase 4 -----------------------------------
__global__ void __launch_bounds__(256)
feats_kernel(const float* __restrict__ hcat, const float* __restrict__ Wout,
             const float* __restrict__ bout, float* __restrict__ feats) {
  const int idx = blockIdx.x * blockDim.x + threadIdx.x;
  if (idx >= T_LEN * TAGS) return;
  const int t = idx / TAGS, k = idx % TAGS;
  const float4* hv = (const float4*)(hcat + (size_t)t * (2 * H2));
  const float4* wv = (const float4*)(Wout + (size_t)k * (2 * H2));
  float acc = bout[k];
#pragma unroll 8
  for (int e = 0; e < (2 * H2) / 4; ++e) {
    float4 h = hv[e], w = wv[e];
    acc = fmaf(h.x, w.x, fmaf(h.y, w.y, fmaf(h.z, w.z, fmaf(h.w, w.w, acc))));
  }
  feats[idx] = acc;
}

// ------------------------------- phase 5 -----------------------------------
__global__ void __launch_bounds__(32)
viterbi_kernel(const float* __restrict__ feats, const float* __restrict__ trans,
               float* __restrict__ out) {
  __shared__ unsigned char bp[T_LEN * TAGS];   // 20 KB of 320 KB LDS
  const int lane = threadIdx.x;
  const bool act = lane < TAGS;

  float fv = (lane == START_TAG) ? 0.0f : NEGV;
  float tr0 = 0, tr1 = 0, tr2 = 0, tr3 = 0, tr4 = 0;
  if (act) {
    tr0 = trans[lane * TAGS + 0];
    tr1 = trans[lane * TAGS + 1];
    tr2 = trans[lane * TAGS + 2];
    tr3 = trans[lane * TAGS + 3];
    tr4 = trans[lane * TAGS + 4];
  }

  for (int t = 0; t < T_LEN; ++t) {
    const float f0 = __shfl(fv, 0), f1 = __shfl(fv, 1), f2 = __shfl(fv, 2),
                f3 = __shfl(fv, 3), f4 = __shfl(fv, 4);
    if (act) {
      float best = f0 + tr0; int b = 0; float v;
      v = f1 + tr1; if (v > best) { best = v; b = 1; }
      v = f2 + tr2; if (v > best) { best = v; b = 2; }
      v = f3 + tr3; if (v > best) { best = v; b = 3; }
      v = f4 + tr4; if (v > best) { best = v; b = 4; }
      fv = best + feats[t * TAGS + lane];
      bp[t * TAGS + lane] = (unsigned char)b;
    }
  }
  __syncthreads();   // single-wave workgroup: effectively free

  const float term = act ? (fv + trans[STOP_TAG * TAGS + lane]) : NEGV;
  const float t0 = __shfl(term, 0), t1 = __shfl(term, 1), t2 = __shfl(term, 2),
              t3 = __shfl(term, 3), t4 = __shfl(term, 4);
  if (lane == 0) {
    int best = 0; float bs = t0;
    if (t1 > bs) { bs = t1; best = 1; }
    if (t2 > bs) { bs = t2; best = 2; }
    if (t3 > bs) { bs = t3; best = 3; }
    if (t4 > bs) { bs = t4; best = 4; }
    out[0] = bs;
    int tag = best;
    for (int t = T_LEN - 1; t >= 0; --t) {
      out[1 + t] = (float)tag;
      tag = bp[t * TAGS + tag];
    }
  }
}

// ------------------------------- launcher ----------------------------------
extern "C" void kernel_launch(void* const* d_in, const int* in_sizes, int n_in,
                              void* d_out, int out_size, void* d_ws, size_t ws_size,
                              hipStream_t stream) {
  (void)in_sizes; (void)n_in; (void)out_size; (void)ws_size;

  const int*   sen   = (const int*)d_in[0];
  const float* emb   = (const float*)d_in[1];
  const float* Wih_f = (const float*)d_in[2];
  const float* Whh_f = (const float*)d_in[3];
  const float* bih_f = (const float*)d_in[4];
  const float* bhh_f = (const float*)d_in[5];
  const float* Wih_b = (const float*)d_in[6];
  const float* Whh_b = (const float*)d_in[7];
  const float* bih_b = (const float*)d_in[8];
  const float* bhh_b = (const float*)d_in[9];
  const float* Wout  = (const float*)d_in[10];
  const float* bout  = (const float*)d_in[11];
  const float* trans = (const float*)d_in[12];
  const float* h0    = (const float*)d_in[13];
  const float* c0    = (const float*)d_in[14];
  float* out = (float*)d_out;

  char* ws = (char*)d_ws;
  size_t off = 0;
  auto alloc = [&](size_t bytes) -> void* {
    void* p = ws + off;
    off = (off + bytes + 255) & ~(size_t)255;
    return p;
  };
  bf16_t* Xbf   = (bf16_t*)alloc((size_t)T_LEN * EMB * sizeof(bf16_t));
  bf16_t* Wbf   = (bf16_t*)alloc((size_t)2 * G4 * EMB * sizeof(bf16_t));
  float*  bsum  = (float*) alloc((size_t)2 * G4 * sizeof(float));
  float*  P     = (float*) alloc((size_t)2 * T_LEN * G4 * sizeof(float));
  float*  hcat  = (float*) alloc((size_t)T_LEN * 2 * H2 * sizeof(float));
  float*  feats = (float*) alloc((size_t)T_LEN * TAGS * sizeof(float));

  // 1) gather + bf16 conversions + bias sums
  prep_kernel<<<T_LEN + 2 * G4 + 8, 256, 0, stream>>>(
      sen, emb, Wih_f, Wih_b, bih_f, bhh_f, bih_b, bhh_b, Xbf, Wbf, bsum);

  // 2) precompute gate inputs with WMMA: P = X @ Wih^T + (bih + bhh)
  input_gemm_kernel<<<dim3(T_LEN / 16, G4 / 16 / 4, 2), 128, 0, stream>>>(
      Xbf, Wbf, bsum, P);

  // 3) both LSTM directions concurrently, Whh persistent in VGPRs (WMMA)
  lstm_wmma_kernel<<<2, 1024, 0, stream>>>(P, Whh_f, Whh_b, h0, c0, hcat);

  // 4) output projection
  feats_kernel<<<(T_LEN * TAGS + 255) / 256, 256, 0, stream>>>(
      hcat, Wout, bout, feats);

  // 5) Viterbi scan + backtrace
  viterbi_kernel<<<1, 32, 0, stream>>>(feats, trans, out);
}
